// ConformerAttention_91207925498343
// MI455X (gfx1250) — matrix operand order repaired and checked by reading, not compiled, not run on hardware
//
#include <hip/hip_runtime.h>

// ---------------------------------------------------------------------------
// Conformer relative-position attention for MI455X (gfx1250, wave32, WMMA).
//  - One elementwise pass converts x / pos_emb / weights to f16 (read once).
//  - All GEMMs: v_wmma_f32_16x16x32_f16, 32x32 tile per wave (2x2 WMMA tiles,
//    each fragment reused twice), 4 wave32 per workgroup.
//  - Attention fused flash-style: bd strip via WMMA + LDS diagonal gather,
//    online softmax with lane shuffles; scores never touch HBM.
//  - Working set (~36 MB f16) is fully L2-resident (192 MB).
// ---------------------------------------------------------------------------

typedef _Float16 v16h __attribute__((ext_vector_type(16)));
typedef _Float16 v8h  __attribute__((ext_vector_type(8)));
typedef float    v8f  __attribute__((ext_vector_type(8)));

#define WMMA(a, b, c) \
    __builtin_amdgcn_wmma_f32_16x16x32_f16(false, (a), false, (b), (short)0, (c), false, false)

#define CB 2
#define CT 2048
#define CD 512
#define CNH 8
#define CDK 64
#define CPP 4096          // pos rows padded (real P = 4095)
#define CSCALE 0.125f     // 1/sqrt(64)

// ---- WMMA fragment loaders (f16 source) -----------------------------------
// A operand (16x32, M x K): lane L: row M = L&15, half g = L>>4.
//   element h -> K = 16*g + h  => one span of 16 consecutive K (32B load).
__device__ __forceinline__ v16h afrag_f16(const _Float16* row, int k0, int g) {
    return *(const v16h*)(row + k0 + (g << 4));
}
// B operand (32x16, K x N): lane L: col N = L&15, half g = L>>4.
//   element h -> K = 16*g + h  => one span of 16 consecutive K (32B load).
__device__ __forceinline__ v16h bfrag_f16(const _Float16* row, int k0, int g) {
    return *(const v16h*)(row + k0 + (g << 4));   // 32B aligned at all call sites
}

// ---------------------------------------------------------------------------
// Kernel 0: f32 -> f16 bulk convert (4 elements / thread)
// ---------------------------------------------------------------------------
__global__ void cvt_f16_kernel(const float* __restrict__ src,
                               _Float16* __restrict__ dst, int n4) {
    const int i = blockIdx.x * blockDim.x + threadIdx.x;
    if (i < n4) {
        float4 v = ((const float4*)src)[i];
        v8h* d = (v8h*)dst;          // store 4 halves = 8B
        _Float16 h0 = (_Float16)v.x, h1 = (_Float16)v.y,
                 h2 = (_Float16)v.z, h3 = (_Float16)v.w;
        ((__attribute__((ext_vector_type(4))) _Float16*)dst)[i] =
            (__attribute__((ext_vector_type(4))) _Float16){h0, h1, h2, h3};
        (void)d;
    }
}

// ---------------------------------------------------------------------------
// Shared 32x32-per-wave f16 GEMM core: C[m,n] = sum_k A[m,k] * B[n,k]
// (B passed as row-major N x K, i.e. C = A @ B^T)
// ---------------------------------------------------------------------------
struct Acc4 { v8f c00, c01, c10, c11; };

__device__ __forceinline__ Acc4 gemm32(const _Float16* __restrict__ A,
                                       const _Float16* __restrict__ Bm,
                                       int lda, int ldb, int m0, int n0, int K,
                                       int nn, int g) {
    const _Float16* ar0 = A  + (size_t)(m0 + nn) * lda;
    const _Float16* ar1 = A  + (size_t)(m0 + 16 + nn) * lda;
    const _Float16* br0 = Bm + (size_t)(n0 + nn) * ldb;
    const _Float16* br1 = Bm + (size_t)(n0 + 16 + nn) * ldb;
    v8f z = {0.f,0.f,0.f,0.f,0.f,0.f,0.f,0.f};
    Acc4 acc = {z, z, z, z};
    for (int k0 = 0; k0 < K; k0 += 32) {
        v16h a0 = afrag_f16(ar0, k0, g);
        v16h a1 = afrag_f16(ar1, k0, g);
        v16h b0 = bfrag_f16(br0, k0, g);
        v16h b1 = bfrag_f16(br1, k0, g);
        acc.c00 = WMMA(a0, b0, acc.c00);
        acc.c01 = WMMA(a0, b1, acc.c01);
        acc.c10 = WMMA(a1, b0, acc.c10);
        acc.c11 = WMMA(a1, b1, acc.c11);
    }
    return acc;
}

// ---------------------------------------------------------------------------
// Kernel 1: qkv = x @ W_qkv^T, scatter into f16 buffers
//   QU/QV: (B,NH,T,DK) = q + pos_bias_u / pos_bias_v
//   KK:    (B,NH,T,DK) row-major
//   VT:    (B,NH,DK,T) transposed (attn@V B-fragments contiguous)
// 128 threads = 4 waves; each wave one 32x32 tile.
// ---------------------------------------------------------------------------
__device__ __forceinline__ void qkv_scatter(int m, int j, float v,
                                            const float* pbu, const float* pbv,
                                            _Float16* QU, _Float16* QV,
                                            _Float16* KK, _Float16* VT) {
    const int b = m >> 11, t = m & (CT - 1);
    if (j < CD) {                                     // Q (+ biases)
        const int h = j >> 6, dd = j & 63;
        const size_t o = (((size_t)(b * CNH + h)) * CT + t) * CDK + dd;
        QU[o] = (_Float16)(v + pbu[j]);
        QV[o] = (_Float16)(v + pbv[j]);
    } else if (j < 2 * CD) {                          // K
        const int jj = j - CD, h = jj >> 6, dd = jj & 63;
        KK[(((size_t)(b * CNH + h)) * CT + t) * CDK + dd] = (_Float16)v;
    } else {                                          // V (transposed)
        const int jj = j - 2 * CD, h = jj >> 6, dd = jj & 63;
        VT[(((size_t)(b * CNH + h)) * CDK + dd) * CT + t] = (_Float16)v;
    }
}

__global__ void qkv_proj_kernel(const _Float16* __restrict__ xh,
                                const _Float16* __restrict__ Wh,
                                const float* __restrict__ pbu,
                                const float* __restrict__ pbv,
                                _Float16* __restrict__ QU, _Float16* __restrict__ QV,
                                _Float16* __restrict__ KK, _Float16* __restrict__ VT) {
    const int wid  = threadIdx.x >> 5;
    const int lane = threadIdx.x & 31;
    const int g = lane >> 4, nn = lane & 15;
    const int n0 = blockIdx.x * 32;                   // [0,1536)
    const int m0 = (blockIdx.y * 4 + wid) * 32;       // [0,4096)
    Acc4 acc = gemm32(xh, Wh, CD, CD, m0, n0, CD, nn, g);
    const int crow = g << 3;
#pragma unroll
    for (int r = 0; r < 8; ++r) {
        qkv_scatter(m0 + crow + r,      n0 + nn,      acc.c00[r], pbu, pbv, QU, QV, KK, VT);
        qkv_scatter(m0 + crow + r,      n0 + 16 + nn, acc.c01[r], pbu, pbv, QU, QV, KK, VT);
        qkv_scatter(m0 + 16 + crow + r, n0 + nn,      acc.c10[r], pbu, pbv, QU, QV, KK, VT);
        qkv_scatter(m0 + 16 + crow + r, n0 + 16 + nn, acc.c11[r], pbu, pbv, QU, QV, KK, VT);
    }
}

// ---------------------------------------------------------------------------
// Kernel 2: pos = pos_emb @ W_pos^T -> POS (NH, 4096, DK) f16 (row 4095 pad)
// pos_emb f16 copy is padded to 4096 rows by the converter (extra row junk,
// clamped here anyway via row 4094 duplication being unnecessary: rows <=4095
// read from the padded buffer are always in-bounds).
// ---------------------------------------------------------------------------
__global__ void pos_proj_kernel(const _Float16* __restrict__ peh,
                                const _Float16* __restrict__ Wh,
                                _Float16* __restrict__ POS) {
    const int wid  = threadIdx.x >> 5;
    const int lane = threadIdx.x & 31;
    const int g = lane >> 4, nn = lane & 15;
    const int n0 = blockIdx.x * 32;                   // [0,512)
    const int m0 = (blockIdx.y * 4 + wid) * 32;       // [0,4096)
    Acc4 acc = gemm32(peh, Wh, CD, CD, m0, n0, CD, nn, g);
    const int crow = g << 3;
#pragma unroll
    for (int r = 0; r < 8; ++r) {
        const int p0 = m0 + crow + r;                 // rows up to 4095 (pad ok)
        const int j0 = n0 + nn;
        POS[((size_t)(j0 >> 6) * CPP + p0) * CDK + (j0 & 63)]                = (_Float16)acc.c00[r];
        POS[((size_t)((j0 + 16) >> 6) * CPP + p0) * CDK + ((j0 + 16) & 63)]  = (_Float16)acc.c01[r];
        POS[((size_t)(j0 >> 6) * CPP + (p0 + 16)) * CDK + (j0 & 63)]         = (_Float16)acc.c10[r];
        POS[((size_t)((j0 + 16) >> 6) * CPP + (p0 + 16)) * CDK + ((j0 + 16) & 63)] = (_Float16)acc.c11[r];
    }
}

// ---------------------------------------------------------------------------
// Kernel 3: fused flash attention with relative-position bd term.
//   scores[t,s] = (qu[t]·k[s] + qv[t]·pos[s-t+T-1]) * scale   (+ mask)
// One wave per (b,h, 16-query-row) tile, streaming s in steps of 32.
// ---------------------------------------------------------------------------
__global__ void attn_kernel(const _Float16* __restrict__ QU, const _Float16* __restrict__ QV,
                            const _Float16* __restrict__ KK, const _Float16* __restrict__ VT,
                            const _Float16* __restrict__ POS,
                            const unsigned char* __restrict__ mask,
                            _Float16* __restrict__ AOUT) {
    const int t0 = blockIdx.x * 16;
    const int bh = blockIdx.y;
    const int b = bh >> 3, h = bh & 7;
    const int lane = threadIdx.x;
    const int g = lane >> 4, nn = lane & 15;
    const int crow = g << 3;                  // C-fragment base row for this lane

    __shared__ __align__(32) float    bd_lds[16][48];
    __shared__ __align__(32) _Float16 p_lds[16][32];

    const _Float16* qu  = QU  + ((size_t)bh * CT) * CDK;
    const _Float16* qv  = QV  + ((size_t)bh * CT) * CDK;
    const _Float16* kk  = KK  + ((size_t)bh * CT) * CDK;
    const _Float16* vt  = VT  + ((size_t)bh * CDK) * CT;
    const _Float16* pos = POS + ((size_t)h  * CPP) * CDK;
    const unsigned char* mrow0 = mask + (size_t)b * CT * CT;

    const _Float16* qur = qu + (size_t)(t0 + nn) * CDK;
    const _Float16* qvr = qv + (size_t)(t0 + nn) * CDK;
    v16h au0 = afrag_f16(qur, 0, g),  au1 = afrag_f16(qur, 32, g);
    v16h av0 = afrag_f16(qvr, 0, g),  av1 = afrag_f16(qvr, 32, g);

    v8f o0 = {0.f,0.f,0.f,0.f,0.f,0.f,0.f,0.f};
    v8f o1 = o0, o2 = o0, o3 = o0;
    float mst[8], lst[8];
    int   allm[8];
#pragma unroll
    for (int r = 0; r < 8; ++r) { mst[r] = -3.0e38f; lst[r] = 0.f; allm[r] = 1; }

    for (int s0 = 0; s0 < CT; s0 += 32) {
        if (s0 + 32 < CT) {
            __builtin_prefetch(kk + (size_t)(s0 + 32 + nn) * CDK, 0, 3);
            __builtin_prefetch(vt + (size_t)nn * CT + (s0 + 32), 0, 3);
        }

        // ---- ac = qu @ k^T : two 16-wide column tiles -----------------
        v8f ac0 = {0.f,0.f,0.f,0.f,0.f,0.f,0.f,0.f}, ac1 = ac0;
        {
            const _Float16* k0r = kk + (size_t)(s0 + nn) * CDK;
            const _Float16* k1r = kk + (size_t)(s0 + 16 + nn) * CDK;
            ac0 = WMMA(au0, bfrag_f16(k0r, 0, g),  ac0);
            ac0 = WMMA(au1, bfrag_f16(k0r, 32, g), ac0);
            ac1 = WMMA(au0, bfrag_f16(k1r, 0, g),  ac1);
            ac1 = WMMA(au1, bfrag_f16(k1r, 32, g), ac1);
        }

        // ---- bd strip: C1[dt, j] = qv[dt] . pos[pbase + j], 48 wide ----
        const int pbase = s0 - t0 + (CT - 1) - 15;      // in [0, 4048]
        v8f c1a = {0.f,0.f,0.f,0.f,0.f,0.f,0.f,0.f}, c1b = c1a, c1c = c1a;
        {
            const _Float16* pa = pos + (size_t)(pbase + nn) * CDK;
            const _Float16* pb = pos + (size_t)(pbase + 16 + nn) * CDK;
            const _Float16* pc = pos + (size_t)(pbase + 32 + nn) * CDK;   // <= row 4095
            c1a = WMMA(av0, bfrag_f16(pa, 0, g),  c1a);
            c1a = WMMA(av1, bfrag_f16(pa, 32, g), c1a);
            c1b = WMMA(av0, bfrag_f16(pb, 0, g),  c1b);
            c1b = WMMA(av1, bfrag_f16(pb, 32, g), c1b);
            c1c = WMMA(av0, bfrag_f16(pc, 0, g),  c1c);
            c1c = WMMA(av1, bfrag_f16(pc, 32, g), c1c);
        }
#pragma unroll
        for (int r = 0; r < 8; ++r) {
            bd_lds[crow + r][nn]      = c1a[r];
            bd_lds[crow + r][16 + nn] = c1b[r];
            bd_lds[crow + r][32 + nn] = c1c[r];
        }
        __syncthreads();

        // ---- scores + online softmax (C-fragment layout) ---------------
#pragma unroll
        for (int r = 0; r < 8; ++r) {
            const int row = crow + r;
            const int tg  = t0 + row;
            const int ds0 = nn, ds1 = 16 + nn;
            const unsigned char mk0 = mrow0[(size_t)tg * CT + (s0 + ds0)];
            const unsigned char mk1 = mrow0[(size_t)tg * CT + (s0 + ds1)];
            const float bd0 = mk0 ? -100000.f : bd_lds[row][15 + ds0 - row] * CSCALE;
            const float bd1 = mk1 ? -100000.f : bd_lds[row][15 + ds1 - row] * CSCALE;
            float sc0 = ac0[r] * CSCALE + bd0;
            float sc1 = ac1[r] * CSCALE + bd1;

            float mt = fmaxf(sc0, sc1);
            mt = fmaxf(mt, __shfl_xor(mt, 8, 32));
            mt = fmaxf(mt, __shfl_xor(mt, 4, 32));
            mt = fmaxf(mt, __shfl_xor(mt, 2, 32));
            mt = fmaxf(mt, __shfl_xor(mt, 1, 32));
            const float nm   = fmaxf(mst[r], mt);
            const float corr = __expf(mst[r] - nm);
            mst[r] = nm;
            const float e0 = __expf(sc0 - nm);
            const float e1 = __expf(sc1 - nm);
            float rs = e0 + e1;
            rs += __shfl_xor(rs, 8, 32);
            rs += __shfl_xor(rs, 4, 32);
            rs += __shfl_xor(rs, 2, 32);
            rs += __shfl_xor(rs, 1, 32);
            lst[r] = lst[r] * corr + rs;
            o0[r] *= corr; o1[r] *= corr; o2[r] *= corr; o3[r] *= corr;
            p_lds[row][ds0] = (_Float16)e0;
            p_lds[row][ds1] = (_Float16)e1;

            int am = (mk0 && mk1) ? 1 : 0;
            am &= __shfl_xor(am, 8, 32);
            am &= __shfl_xor(am, 4, 32);
            am &= __shfl_xor(am, 2, 32);
            am &= __shfl_xor(am, 1, 32);
            allm[r] &= am;
        }
        __syncthreads();

        // ---- out += p @ V : four 16-wide d tiles -----------------------
        v16h pa = afrag_f16(&p_lds[nn][0], 0, g);
        o0 = WMMA(pa, bfrag_f16(vt + (size_t)(0  + nn) * CT + s0, 0, g), o0);
        o1 = WMMA(pa, bfrag_f16(vt + (size_t)(16 + nn) * CT + s0, 0, g), o1);
        o2 = WMMA(pa, bfrag_f16(vt + (size_t)(32 + nn) * CT + s0, 0, g), o2);
        o3 = WMMA(pa, bfrag_f16(vt + (size_t)(48 + nn) * CT + s0, 0, g), o3);
        __syncthreads();
    }

#pragma unroll
    for (int r = 0; r < 8; ++r) {
        float inv = (lst[r] > 0.f) ? (1.0f / lst[r]) : 0.f;
        if (allm[r]) inv = 0.f;
        const int t = t0 + crow + r;
        _Float16* orow = AOUT + ((size_t)b * CT + t) * CD + h * CDK;
        orow[0  + nn] = (_Float16)(o0[r] * inv);
        orow[16 + nn] = (_Float16)(o1[r] * inv);
        orow[32 + nn] = (_Float16)(o2[r] * inv);
        orow[48 + nn] = (_Float16)(o3[r] * inv);
    }
}

// ---------------------------------------------------------------------------
// Kernel 4: out = AOUT(f16) @ W_out^T(f16) -> d_out (f32)
// ---------------------------------------------------------------------------
__global__ void out_proj_kernel(const _Float16* __restrict__ A,
                                const _Float16* __restrict__ Wh,
                                float* __restrict__ out) {
    const int wid  = threadIdx.x >> 5;
    const int lane = threadIdx.x & 31;
    const int g = lane >> 4, nn = lane & 15;
    const int n0 = blockIdx.x * 32;                   // [0,512)
    const int m0 = (blockIdx.y * 4 + wid) * 32;       // [0,4096)
    Acc4 acc = gemm32(A, Wh, CD, CD, m0, n0, CD, nn, g);
    const int crow = g << 3;
#pragma unroll
    for (int r = 0; r < 8; ++r) {
        out[(size_t)(m0 + crow + r)      * CD + (n0 + nn)]      = acc.c00[r];
        out[(size_t)(m0 + crow + r)      * CD + (n0 + 16 + nn)] = acc.c01[r];
        out[(size_t)(m0 + 16 + crow + r) * CD + (n0 + nn)]      = acc.c10[r];
        out[(size_t)(m0 + 16 + crow + r) * CD + (n0 + 16 + nn)] = acc.c11[r];
    }
}

// ---------------------------------------------------------------------------
extern "C" void kernel_launch(void* const* d_in, const int* in_sizes, int n_in,
                              void* d_out, int out_size, void* d_ws, size_t ws_size,
                              hipStream_t stream) {
    const float* x    = (const float*)d_in[0];              // (B,T,D)
    const float* pe   = (const float*)d_in[1];              // (1,P,D)
    const unsigned char* msk = (const unsigned char*)d_in[2]; // (B,T,T) bool
    const float* Wqkv = (const float*)d_in[3];              // (3D,D)
    const float* Wpos = (const float*)d_in[4];              // (D,D)
    const float* Wout = (const float*)d_in[5];              // (D,D)
    const float* pbu  = (const float*)d_in[6];              // (NH,DK)
    const float* pbv  = (const float*)d_in[7];              // (NH,DK)
    float* out = (float*)d_out;

    // Workspace carve-up (halves). Region sizes all multiples of 32B.
    const size_t SZ = (size_t)CB * CNH * CT * CDK;          // 2,097,152 elems
    _Float16* p    = (_Float16*)d_ws;
    _Float16* QU   = p;            p += SZ;
    _Float16* QV   = p;            p += SZ;
    _Float16* KK   = p;            p += SZ;
    _Float16* VT   = p;            p += SZ;
    _Float16* POS  = p;            p += SZ;                 // NH*4096*DK == SZ
    _Float16* AOUT = p;            p += SZ;
    _Float16* XH   = p;            p += (size_t)4096 * CD;  // 2,097,152
    _Float16* PEH  = p;            p += (size_t)4096 * CD;  // 4095 rows used + pad
    _Float16* WQKVH= p;            p += (size_t)1536 * CD;
    _Float16* WPOSH= p;            p += (size_t)CD * CD;
    _Float16* WOUTH= p;            p += (size_t)CD * CD;

    dim3 cb(256, 1, 1);
    // bulk f32 -> f16 conversions (each element read exactly once afterwards)
    {
        int n4;
        n4 = (4096 * CD) / 4;     cvt_f16_kernel<<<(n4 + 255) / 256, cb, 0, stream>>>(x,    XH,    n4);
        n4 = (4095 * CD) / 4;     cvt_f16_kernel<<<(n4 + 255) / 256, cb, 0, stream>>>(pe,   PEH,   n4);
        n4 = (1536 * CD) / 4;     cvt_f16_kernel<<<(n4 + 255) / 256, cb, 0, stream>>>(Wqkv, WQKVH, n4);
        n4 = (CD * CD) / 4;       cvt_f16_kernel<<<(n4 + 255) / 256, cb, 0, stream>>>(Wpos, WPOSH, n4);
        n4 = (CD * CD) / 4;       cvt_f16_kernel<<<(n4 + 255) / 256, cb, 0, stream>>>(Wout, WOUTH, n4);
        // pad row 4095 of PEH (read by pos_proj tile 255, value irrelevant but
        // must be deterministic): reuse converter on row 4094 duplicated.
        n4 = CD / 4;              cvt_f16_kernel<<<(n4 + 255) / 256, cb, 0, stream>>>(pe + (size_t)4094 * CD, PEH + (size_t)4095 * CD, n4);
    }

    dim3 gblk(128, 1, 1);                                   // 4 wave32 / WG
    qkv_proj_kernel<<<dim3(48, 32), gblk, 0, stream>>>(XH, WQKVH, pbu, pbv, QU, QV, KK, VT);
    pos_proj_kernel<<<dim3(16, 32), gblk, 0, stream>>>(PEH, WPOSH, POS);
    attn_kernel    <<<dim3(128, 16), dim3(32,1,1), 0, stream>>>(QU, QV, KK, VT, POS, msk, AOUT);
    out_proj_kernel<<<dim3(16, 32), gblk, 0, stream>>>(AOUT, WOUTH, out);
}
